// TannerGNN_65592740544941
// MI455X (gfx1250) — matrix operand order repaired and checked by reading, not compiled, not run on hardware
//
#include <hip/hip_runtime.h>
#include <hip/hip_bf16.h>
#include <cstdint>
#include <cstddef>

#define H 128
#define TYPES 2

// ---------------------------------------------------------------------------
// WMMA types / helpers (CDNA5 gfx1250, wave32)
// ---------------------------------------------------------------------------
typedef __attribute__((ext_vector_type(16))) __bf16 bf16x16;
typedef __attribute__((ext_vector_type(8)))  float  f32x8;

struct alignas(16) U4 { unsigned int x, y, z, w; };
union Frag16 { bf16x16 v; U4 q[2]; unsigned short s[16]; };

__device__ __forceinline__ unsigned short f2bf(float f) {
    unsigned int u = __float_as_uint(f);
    u += 0x7fffu + ((u >> 16) & 1u);       // round-to-nearest-even
    return (unsigned short)(u >> 16);
}

// A-fragment (16x32 bf16): lane<16 holds row M=lane, K = kbase+[0..7] and kbase+16+[0..7];
// lane>=16 holds same row, K offset +8 (per ISA 7.12.2 layout).
__device__ __forceinline__ bf16x16 afrag_bf16(const unsigned short* row, int kbase, int hi) {
    int k0 = kbase + (hi << 3);
    Frag16 r;
    r.q[0] = *(const U4*)(row + k0);
    r.q[1] = *(const U4*)(row + k0 + 16);
    return r.v;
}

__device__ __forceinline__ bf16x16 afrag_f32(const float* row, int kbase, int hi) {
    int k0 = kbase + (hi << 3);
    Frag16 r;
#pragma unroll
    for (int j = 0; j < 8; ++j) {
        r.s[j]     = f2bf(row[k0 + j]);
        r.s[8 + j] = f2bf(row[k0 + 16 + j]);
    }
    return r.v;
}

// B-fragment (32x16 bf16): lane<16 -> col N=lane, K = kbase+[0..7] and kbase+16+[0..7];
// lane>=16 -> K offset +8 (matches A layout).  Wt is [N][K] (K contiguous).
__device__ __forceinline__ bf16x16 bfrag(const unsigned short* Wt, int ncol, int K,
                                         int kbase, int hi) {
    const unsigned short* p = Wt + (size_t)ncol * K + kbase + (hi << 3);
    Frag16 r;
    r.q[0] = *(const U4*)p;
    r.q[1] = *(const U4*)(p + 16);
    return r.v;
}

__device__ __forceinline__ f32x8 wmma_bf16(bf16x16 a, bf16x16 b, f32x8 c) {
    return __builtin_amdgcn_wmma_f32_16x16x32_bf16(false, a, false, b, (short)0, c,
                                                   false, false);
}

__device__ __forceinline__ float sigm(float x) { return 1.0f / (1.0f + __expf(-x)); }

#define ZERO8 (f32x8){0.f,0.f,0.f,0.f,0.f,0.f,0.f,0.f}

// ---------------------------------------------------------------------------
// Small prep kernels
// ---------------------------------------------------------------------------
__global__ void k_input(const float* __restrict__ x, const float* __restrict__ Win,
                        const float* __restrict__ bin, float* __restrict__ h, int total) {
    int tid = blockIdx.x * blockDim.x + threadIdx.x;
    if (tid >= total) return;
    int node = tid >> 7, j = tid & 127;
    const float* xr = x + (size_t)node * 4;
    float s = bin[j];
#pragma unroll
    for (int k = 0; k < 4; ++k) s += xr[k] * Win[k * H + j];
    h[tid] = s > 0.f ? s : 0.f;
}

__global__ void k_f2bf(const float* __restrict__ s, unsigned short* __restrict__ d, int n) {
    int tid = blockIdx.x * blockDim.x + threadIdx.x;
    if (tid < n) d[tid] = f2bf(s[tid]);
}

__global__ void k_zero(float* __restrict__ p, int n) {
    int tid = blockIdx.x * blockDim.x + threadIdx.x;
    if (tid < n) p[tid] = 0.f;
}

// src: [nmat][K][N] f32 -> dst: [nmat][N][K] bf16  (K becomes contiguous per column)
__global__ void k_tr(const float* __restrict__ src, unsigned short* __restrict__ dst,
                     int K, int N, int total) {
    int tid = blockIdx.x * blockDim.x + threadIdx.x;
    if (tid >= total) return;
    int kn = K * N;
    int m = tid / kn, rem = tid - m * kn;
    int k = rem / N, n = rem - k * N;
    dst[(size_t)m * kn + (size_t)n * K + k] = f2bf(src[tid]);
}

// ---------------------------------------------------------------------------
// Edge message MLP + scatter-add (the ~236 GFLOP hot loop)
// one wave = 16 edges;   cat[256] -> relu(GEMM1) -> GEMM2 -> masked atomics
// ---------------------------------------------------------------------------
__global__ __launch_bounds__(128) void k_edge(
        const unsigned short* __restrict__ hbf,
        const int* __restrict__ src, const int* __restrict__ dst,
        const int* __restrict__ etype,
        const unsigned short* __restrict__ W1t, const float* __restrict__ b1,
        const unsigned short* __restrict__ W2t, const float* __restrict__ b2,
        float* __restrict__ agg, int n_edges) {
    __shared__ __align__(16) unsigned short relu_s[4][16 * H];   // 4 KB / wave

    const int lane  = threadIdx.x & 31;
    const int wid   = threadIdx.x >> 5;
    const int laneM = lane & 15;
    const int hi    = lane >> 4;
    const int e0    = (blockIdx.x * 4 + wid) * 16;

    int eM = e0 + laneM;
    if (eM >= n_edges) eM = n_edges - 1;          // clamp; masked at the atomic
    __builtin_prefetch(etype + (eM < n_edges - 16 ? eM + 16 : eM), 0, 0);

    const unsigned short* rowS = hbf + (size_t)src[eM] * H;
    const unsigned short* rowD = hbf + (size_t)dst[eM] * H;

    // cat = [h[src], h[dst]] : K-steps 0..3 come from src row, 4..7 from dst row
    bf16x16 A[8];
#pragma unroll
    for (int ks = 0; ks < 4; ++ks) {
        A[ks]     = afrag_bf16(rowS, ks * 32, hi);
        A[ks + 4] = afrag_bf16(rowD, ks * 32, hi);
    }

    for (int t = 0; t < TYPES; ++t) {
        const unsigned short* W1p = W1t + (size_t)t * H * (2 * H);
        const unsigned short* W2p = W2t + (size_t)t * H * H;

        // ---- GEMM1: [16x256] @ [256x128] + b1, relu, stage to LDS ----
#pragma unroll
        for (int nt = 0; nt < 8; ++nt) {
            f32x8 acc = ZERO8;
            const int ncol = nt * 16 + laneM;
#pragma unroll
            for (int ks = 0; ks < 8; ++ks)
                acc = wmma_bf16(A[ks], bfrag(W1p, ncol, 2 * H, ks * 32, hi), acc);
            const float bias = b1[t * H + ncol];
#pragma unroll
            for (int i = 0; i < 8; ++i) {
                float v = acc[i] + bias;
                v = v > 0.f ? v : 0.f;
                relu_s[wid][(i + 8 * hi) * H + ncol] = f2bf(v);   // C-layout -> row-major
            }
        }
        __syncthreads();

        // ---- GEMM2: relu-tile [16x128] @ [128x128] + b2, masked scatter-add ----
        bf16x16 A2[4];
#pragma unroll
        for (int ks = 0; ks < 4; ++ks)
            A2[ks] = afrag_bf16(&relu_s[wid][laneM * H], ks * 32, hi);

#pragma unroll
        for (int nt = 0; nt < 8; ++nt) {
            f32x8 acc = ZERO8;
            const int ncol = nt * 16 + laneM;
#pragma unroll
            for (int ks = 0; ks < 4; ++ks)
                acc = wmma_bf16(A2[ks], bfrag(W2p, ncol, H, ks * 32, hi), acc);
            const float bias = b2[t * H + ncol];
#pragma unroll
            for (int i = 0; i < 8; ++i) {
                const int M = i + 8 * hi;
                const int e = e0 + M;
                if (e < n_edges && etype[e] == t)
                    atomicAdd(&agg[(size_t)dst[e] * H + ncol], acc[i] + bias);
            }
        }
        __syncthreads();
    }
}

// ---------------------------------------------------------------------------
// GRU cell: gi = agg@W_ih+b_ih ; gh = h@W_hh+b_hh ; gate math in fp32
// one wave = 16 nodes, N=384 split into 8 feature tiles x 3 gates
// ---------------------------------------------------------------------------
__global__ __launch_bounds__(128) void k_gru(
        const float* __restrict__ agg, const float* __restrict__ h,
        const unsigned short* __restrict__ hbf,
        const unsigned short* __restrict__ Wiht, const float* __restrict__ bih,
        const unsigned short* __restrict__ Whht, const float* __restrict__ bhh,
        float* __restrict__ hnew, int n_nodes) {
    const int lane  = threadIdx.x & 31;
    const int laneM = lane & 15;
    const int hi    = lane >> 4;
    const int m0    = (blockIdx.x * 4 + (threadIdx.x >> 5)) * 16;
    if (m0 >= n_nodes) return;                       // wave-uniform, no barriers here

    int mM = m0 + laneM;
    if (mM >= n_nodes) mM = n_nodes - 1;
    const float*          arow = agg + (size_t)mM * H;
    const unsigned short* hrow = hbf + (size_t)mM * H;

    bf16x16 Aa[4], Ah[4];
#pragma unroll
    for (int ks = 0; ks < 4; ++ks) {
        Aa[ks] = afrag_f32 (arow, ks * 32, hi);      // agg is fp32 (atomic target)
        Ah[ks] = afrag_bf16(hrow, ks * 32, hi);
    }

#pragma unroll
    for (int f = 0; f < 8; ++f) {
        f32x8 ir = ZERO8, iz = ZERO8, in_ = ZERO8, hr = ZERO8, hz = ZERO8, hn = ZERO8;
        const int nr = f * 16 + laneM;
#pragma unroll
        for (int ks = 0; ks < 4; ++ks) {
            const int kb = ks * 32;
            ir  = wmma_bf16(Aa[ks], bfrag(Wiht, nr,       H, kb, hi), ir);
            iz  = wmma_bf16(Aa[ks], bfrag(Wiht, nr + 128, H, kb, hi), iz);
            in_ = wmma_bf16(Aa[ks], bfrag(Wiht, nr + 256, H, kb, hi), in_);
            hr  = wmma_bf16(Ah[ks], bfrag(Whht, nr,       H, kb, hi), hr);
            hz  = wmma_bf16(Ah[ks], bfrag(Whht, nr + 128, H, kb, hi), hz);
            hn  = wmma_bf16(Ah[ks], bfrag(Whht, nr + 256, H, kb, hi), hn);
        }
        const float bir = bih[nr], biz = bih[nr + 128], bin = bih[nr + 256];
        const float bhr = bhh[nr], bhz = bhh[nr + 128], bhn = bhh[nr + 256];
#pragma unroll
        for (int i = 0; i < 8; ++i) {
            const int M = i + 8 * hi;
            const int node = m0 + M;
            const int nc = node < n_nodes ? node : n_nodes - 1;
            const float hv = h[(size_t)nc * H + nr];
            const float r = sigm(ir[i] + bir + hr[i] + bhr);
            const float z = sigm(iz[i] + biz + hz[i] + bhz);
            const float n = tanhf(in_[i] + bin + r * (hn[i] + bhn));
            const float o = (1.f - z) * n + z * hv;
            if (node < n_nodes) hnew[(size_t)node * H + nr] = o;
        }
    }
}

// ---------------------------------------------------------------------------
// Readout: relu(h[:nd]@Wr1+br1)@Wr2+br2 ; WMMA then shfl_xor dot-reduce
// ---------------------------------------------------------------------------
__global__ __launch_bounds__(128) void k_readout(
        const unsigned short* __restrict__ hbf,
        const unsigned short* __restrict__ Wr1t, const float* __restrict__ br1,
        const float* __restrict__ Wr2, const float* __restrict__ br2,
        const int* __restrict__ ndp, float* __restrict__ out) {
    const int nd = *ndp;                              // n_data lives in device memory
    const int lane  = threadIdx.x & 31;
    const int laneM = lane & 15;
    const int hi    = lane >> 4;
    const int m0    = (blockIdx.x * 4 + (threadIdx.x >> 5)) * 16;
    if (m0 >= nd) return;                             // wave-uniform

    int mM = m0 + laneM;
    if (mM >= nd) mM = nd - 1;
    const unsigned short* hrow = hbf + (size_t)mM * H;

    bf16x16 A[4];
#pragma unroll
    for (int ks = 0; ks < 4; ++ks) A[ks] = afrag_bf16(hrow, ks * 32, hi);

    float psum[8] = {0.f, 0.f, 0.f, 0.f, 0.f, 0.f, 0.f, 0.f};
#pragma unroll
    for (int nt = 0; nt < 8; ++nt) {
        f32x8 acc = ZERO8;
        const int ncol = nt * 16 + laneM;
#pragma unroll
        for (int ks = 0; ks < 4; ++ks)
            acc = wmma_bf16(A[ks], bfrag(Wr1t, ncol, H, ks * 32, hi), acc);
        const float bias = br1[ncol];
        const float w    = Wr2[ncol];
#pragma unroll
        for (int i = 0; i < 8; ++i) {
            float v = acc[i] + bias;
            v = v > 0.f ? v : 0.f;
            psum[i] += v * w;                        // partial of row (i+8*hi) dot Wr2
        }
    }
    // butterfly reduce across the 16 lanes holding each row's N-slices
#pragma unroll
    for (int d = 1; d < 16; d <<= 1)
#pragma unroll
        for (int i = 0; i < 8; ++i) psum[i] += __shfl_xor(psum[i], d, 32);

    if (laneM == 0) {
        const float b = br2[0];
#pragma unroll
        for (int i = 0; i < 8; ++i) {
            const int node = m0 + i + 8 * hi;
            if (node < nd) out[node] = psum[i] + b;
        }
    }
}

// ---------------------------------------------------------------------------
// Host orchestration (graph-capture safe: kernels only, all on `stream`)
// ---------------------------------------------------------------------------
extern "C" void kernel_launch(void* const* d_in, const int* in_sizes, int n_in,
                              void* d_out, int out_size, void* d_ws, size_t ws_size,
                              hipStream_t stream) {
    const float* x    = (const float*)d_in[0];
    const int*   eidx = (const int*)  d_in[1];
    const int*   etyp = (const int*)  d_in[2];
    const int*   ndp  = (const int*)  d_in[3];
    const float* Win  = (const float*)d_in[4];
    const float* bin  = (const float*)d_in[5];
    const float* W1   = (const float*)d_in[6];
    const float* b1   = (const float*)d_in[7];
    const float* W2   = (const float*)d_in[8];
    const float* b2   = (const float*)d_in[9];
    const float* Wih  = (const float*)d_in[10];
    const float* bih  = (const float*)d_in[11];
    const float* Whh  = (const float*)d_in[12];
    const float* bhh  = (const float*)d_in[13];
    const float* Wr1  = (const float*)d_in[14];
    const float* br1  = (const float*)d_in[15];
    const float* Wr2  = (const float*)d_in[16];
    const float* br2  = (const float*)d_in[17];

    const int n_nodes = in_sizes[0] / 4;          // FEAT = 4
    const int n_edges = in_sizes[2];
    const int L       = in_sizes[7] / (TYPES * H); // b1 = [L,T,H]
    const int nh      = n_nodes * H;

    // --- carve workspace ---
    size_t off = 0;
    auto carve = [&](size_t bytes) -> void* {
        void* p = (char*)d_ws + off;
        off += (bytes + 255) & ~(size_t)255;
        return p;
    };
    float*          h_a  = (float*)carve((size_t)nh * 4);
    float*          h_b  = (float*)carve((size_t)nh * 4);
    unsigned short* hbf  = (unsigned short*)carve((size_t)nh * 2);
    float*          agg  = (float*)carve((size_t)nh * 4);
    unsigned short* W1t  = (unsigned short*)carve((size_t)L * TYPES * 2 * H * H * 2);
    unsigned short* W2t  = (unsigned short*)carve((size_t)L * TYPES * H * H * 2);
    unsigned short* Wiht = (unsigned short*)carve((size_t)L * 3 * H * H * 2);
    unsigned short* Whht = (unsigned short*)carve((size_t)L * 3 * H * H * 2);
    unsigned short* Wr1t = (unsigned short*)carve((size_t)H * H * 2);
    (void)ws_size; (void)n_in;

    const int TB = 256;
    auto blks = [](int total, int tb) { return (total + tb - 1) / tb; };

    // --- weight transpose+convert (deterministic, rerun every call) ---
    {
        int t1 = L * TYPES * 2 * H * H;
        k_tr<<<blks(t1, TB), TB, 0, stream>>>(W1, W1t, 2 * H, H, t1);
        int t2 = L * TYPES * H * H;
        k_tr<<<blks(t2, TB), TB, 0, stream>>>(W2, W2t, H, H, t2);
        int t3 = L * H * 3 * H;
        k_tr<<<blks(t3, TB), TB, 0, stream>>>(Wih, Wiht, H, 3 * H, t3);
        k_tr<<<blks(t3, TB), TB, 0, stream>>>(Whh, Whht, H, 3 * H, t3);
        int t4 = H * H;
        k_tr<<<blks(t4, TB), TB, 0, stream>>>(Wr1, Wr1t, H, H, t4);
    }

    // --- h0 = relu(x @ W_in + b_in) ---
    k_input<<<blks(nh, TB), TB, 0, stream>>>(x, Win, bin, h_a, nh);

    const int* src = eidx;
    const int* dst = eidx + n_edges;
    const int edge_blocks = (((n_edges + 15) / 16) + 3) / 4;
    const int node_blocks = (((n_nodes + 15) / 16) + 3) / 4;

    float* h_cur = h_a;
    float* h_nxt = h_b;
    for (int l = 0; l < L; ++l) {
        k_f2bf<<<blks(nh, TB), TB, 0, stream>>>(h_cur, hbf, nh);
        k_zero<<<blks(nh, TB), TB, 0, stream>>>(agg, nh);
        k_edge<<<edge_blocks, 128, 0, stream>>>(
            hbf, src, dst, etyp,
            W1t + (size_t)l * TYPES * 2 * H * H, b1 + (size_t)l * TYPES * H,
            W2t + (size_t)l * TYPES * H * H,     b2 + (size_t)l * TYPES * H,
            agg, n_edges);
        k_gru<<<node_blocks, 128, 0, stream>>>(
            agg, h_cur, hbf,
            Wiht + (size_t)l * 3 * H * H, bih + (size_t)l * 3 * H,
            Whht + (size_t)l * 3 * H * H, bhh + (size_t)l * 3 * H,
            h_nxt, n_nodes);
        float* tmp = h_cur; h_cur = h_nxt; h_nxt = tmp;
    }

    // --- readout ---
    k_f2bf<<<blks(nh, TB), TB, 0, stream>>>(h_cur, hbf, nh);
    k_readout<<<node_blocks, 128, 0, stream>>>(hbf, Wr1t, br1, Wr2, br2, ndp,
                                               (float*)d_out);
    (void)out_size;
}
